// MPNNFlockingModel_23158463660136
// MI455X (gfx1250) — matrix operand order repaired and checked by reading, not compiled, not run on hardware
//
#include <hip/hip_runtime.h>

// ---------------------------------------------------------------------------
// MPNN flocking model for gfx1250 (MI455X).
// Strategy: bf16 activations/weights through V_WMMA_F32_16X16X32_BF16
// (f32 accumulate), f32 BatchNorm statistics via global atomics,
// fused bias / BN / ReLU / scatter stages. Wave32 throughout.
// All address arithmetic kept in unsigned 32-bit offsets against uniform
// SGPR base pointers so loads/stores lower to saddr-form (GVS) VMEM with
// immediate offsets instead of per-lane 64-bit address chains.
// ---------------------------------------------------------------------------

typedef unsigned short u16;
typedef unsigned int   u32;
typedef __attribute__((ext_vector_type(16))) __bf16 bf16x16;
typedef __attribute__((ext_vector_type(8)))  float  floatx8;

union BFrag {
  u16     u[16];
  uint4   q[2];
  bf16x16 v;
};

__device__ __forceinline__ u16 f2bf(float f) {
  unsigned int u = __builtin_bit_cast(unsigned int, f);
  u += 0x7FFFu + ((u >> 16) & 1u);           // round-to-nearest-even
  return (u16)(u >> 16);
}

__device__ __forceinline__ floatx8 wmma_bf16(bf16x16 a, bf16x16 b, floatx8 c) {
  // (neg_a, A, neg_b, B, c_mod, C, reuse_a, reuse_b)
  return __builtin_amdgcn_wmma_f32_16x16x32_bf16(false, a, false, b,
                                                 (short)0, c, false, false);
}

// ---------------------------------------------------------------------------
// f32 -> bf16 buffer convert (weights)
// ---------------------------------------------------------------------------
__global__ void k_f2bf(const float* __restrict__ in, u16* __restrict__ out, int n) {
  int i = blockIdx.x * blockDim.x + threadIdx.x;
  if (i < n) out[i] = f2bf(in[i]);
}

// ---------------------------------------------------------------------------
// h = [pos, vel] @ W_in + b_in  -> bf16, written to xu[:, 0:64] (stride 128)
// ---------------------------------------------------------------------------
__global__ void k_embed(const float* __restrict__ pos, const float* __restrict__ vel,
                        const float* __restrict__ Win, const float* __restrict__ bin,
                        u16* __restrict__ xu, int N) {
  int i = blockIdx.x * blockDim.x + threadIdx.x;
  if (i >= N * 64) return;
  int n = i >> 6, c = i & 63;
  float h = bin[c]
          + pos[2 * n]     * Win[0 * 64 + c]
          + pos[2 * n + 1] * Win[1 * 64 + c]
          + vel[2 * n]     * Win[2 * 64 + c]
          + vel[2 * n + 1] * Win[3 * 64 + c];
  xu[(size_t)n * 128 + c] = f2bf(h);
}

// ---------------------------------------------------------------------------
// Persistent WMMA GEMM:  Z[rows,64] = A[rows,KDIM](bf16) @ W[KDIM,64](bf16) + bias
// Also accumulates per-channel sum(z), sum(z^2) into stats[0:64], stats[64:128].
// GATHER mode: logical A row e = concat(h[dst[e]], h[src[e]]) read from xu
// (node rows of 128 bf16).  Contiguous mode: A rows of KDIM bf16.
//
// Fragment layouts (contiguous K order used by this kernel for A and B):
//  A (16x32 bf16): lanes m and m+16 hold row m; lane m   -> K 0..15
//                                                lane m+16-> K 16..31
//  B (32x16 bf16): lane L holds column L&15;   K = (L>=16?16:0)+0..15
//  C/D (16x16 f32): VGPR r of lane L -> row r + (L>=16?8:0), col L&15
// ---------------------------------------------------------------------------
template<int KDIM, bool GATHER>
__global__ void __launch_bounds__(256)
k_gemm_bn(const u16* __restrict__ A,
          const int* __restrict__ esrc, const int* __restrict__ edst,
          const u16* __restrict__ Wb, const float* __restrict__ bias,
          float* __restrict__ Z, float* __restrict__ stats, int ntiles) {
  constexpr int KS = KDIM / 32;
  const int lane = threadIdx.x & 31;
  const int m    = lane & 15;           // row within tile / output column
  const int half = lane >> 4;
  const int wave  = blockIdx.x * (blockDim.x >> 5) + (threadIdx.x >> 5);
  const int nwav  = gridDim.x * (blockDim.x >> 5);

  const char* Ab = (const char*)A;
  char*       Zb = (char*)Z;

  // Preload all B fragments (held in VGPRs for the whole kernel).
  bf16x16 bfrag[KS][4];
#pragma unroll
  for (int ks = 0; ks < KS; ++ks) {
#pragma unroll
    for (int t = 0; t < 4; ++t) {
      BFrag f;
      const int kbase = ks * 32 + half * 16;
      const int c     = t * 16 + m;
#pragma unroll
      for (int i = 0; i < 16; ++i)
        f.u[i] = Wb[(size_t)(kbase + i) * 64 + c];
      bfrag[ks][t] = f.v;
    }
  }

  float bcol[4];
#pragma unroll
  for (int t = 0; t < 4; ++t) bcol[t] = bias[t * 16 + m];

  float s1[4] = {0.f, 0.f, 0.f, 0.f};
  float s2[4] = {0.f, 0.f, 0.f, 0.f};
  const floatx8 vzero = {0.f, 0.f, 0.f, 0.f, 0.f, 0.f, 0.f, 0.f};

  for (int tile = wave; tile < ntiles; tile += nwav) {
    const int row = tile * 16 + m;
    u32 nd = 0, ns = 0;               // byte offsets of the two source rows
    if (GATHER) {
      nd = (u32)edst[row] << 8;       // node row: 128 bf16 = 256 B
      ns = (u32)esrc[row] << 8;
    } else {
      nd = (u32)row * (u32)(KDIM * 2);
    }

    floatx8 acc[4] = {vzero, vzero, vzero, vzero};
#pragma unroll
    for (int ks = 0; ks < KS; ++ks) {
      const int kk = ks * 32;
      const int c0 = kk + half * 16;
      const int c1 = kk + half * 16 + 8;
      BFrag f;
      if (GATHER) {
        const u32 o0 = ((c0 < 64) ? nd : ns) + (u32)((c0 & 63) * 2);
        const u32 o1 = ((c1 < 64) ? nd : ns) + (u32)((c1 & 63) * 2);
        f.q[0] = *(const uint4*)(Ab + o0);
        f.q[1] = *(const uint4*)(Ab + o1);
      } else {
        f.q[0] = *(const uint4*)(Ab + nd + (u32)(c0 * 2));
        f.q[1] = *(const uint4*)(Ab + nd + (u32)(c1 * 2));
      }
#pragma unroll
      for (int t = 0; t < 4; ++t)
        acc[t] = wmma_bf16(f.v, bfrag[ks][t], acc[t]);
    }

    // Store 4 x (8 rows) with one 32-bit base per fragment column and
    // constant row offsets (r * 256 B) absorbed into the VMEM IOFFSET.
    const u32 zbase = (u32)(tile * 16 + half * 8) * 256u + (u32)m * 4u;
#pragma unroll
    for (int t = 0; t < 4; ++t) {
      float* zp = (float*)(Zb + zbase + (u32)t * 64u);
#pragma unroll
      for (int r = 0; r < 8; ++r) {
        const float z = acc[t][r] + bcol[t];
        zp[r * 64] = z;
        s1[t] += z;
        s2[t] += z * z;
      }
    }
  }

#pragma unroll
  for (int t = 0; t < 4; ++t) {
    atomicAdd(&stats[t * 16 + m], s1[t]);
    atomicAdd(&stats[64 + t * 16 + m], s2[t]);
  }
}

// ---------------------------------------------------------------------------
// Fold BN batch statistics into per-channel scale/shift:
//   scale = gamma * rsqrt(var+eps), shift = beta - mu*scale
// ---------------------------------------------------------------------------
__global__ void k_bn_finalize(float* __restrict__ stats,
                              const float* __restrict__ gamma,
                              const float* __restrict__ beta, float inv_count) {
  int c = threadIdx.x;
  if (c < 64) {
    float mu  = stats[c] * inv_count;
    float var = stats[64 + c] * inv_count - mu * mu;
    float sc  = gamma[c] * rsqrtf(var + 1e-5f);
    stats[128 + c] = sc;
    stats[192 + c] = beta[c] - mu * sc;
  }
}

// y = relu(z*scale+shift) -> bf16
__global__ void k_bnrelu_bf16(const float* __restrict__ Z,
                              const float* __restrict__ stats,
                              u16* __restrict__ out, size_t n) {
  size_t i = (size_t)blockIdx.x * blockDim.x + threadIdx.x;
  if (i >= n) return;
  int c = (int)(i & 63);
  out[i] = f2bf(fmaxf(Z[i] * stats[128 + c] + stats[192 + c], 0.f));
}

// msg = relu(z*scale+shift); aggr[dst[e]] += msg  (segment-sum via f32 atomics)
__global__ void k_bnrelu_scatter(const float* __restrict__ Z,
                                 const float* __restrict__ stats,
                                 const int* __restrict__ edst,
                                 float* __restrict__ aggr, size_t n) {
  size_t i = (size_t)blockIdx.x * blockDim.x + threadIdx.x;
  if (i >= n) return;
  int c = (int)(i & 63);
  size_t e = i >> 6;
  float y = fmaxf(Z[i] * stats[128 + c] + stats[192 + c], 0.f);
  atomicAdd(&aggr[((u32)edst[e] << 6) + (u32)c], y);
}

// xu[:, 64:128] = bf16(aggr)
__global__ void k_aggr2xu(const float* __restrict__ aggr, u16* __restrict__ xu, int n64) {
  int i = blockIdx.x * blockDim.x + threadIdx.x;
  if (i >= n64) return;
  int n = i >> 6, c = i & 63;
  xu[(size_t)n * 128 + 64 + c] = f2bf(aggr[i]);
}

// out[n] = relu(zn*scale+shift) . W_pred + b_pred
__global__ void k_pred(const float* __restrict__ zn, const float* __restrict__ stats,
                       const float* __restrict__ Wp, const float* __restrict__ bp,
                       float* __restrict__ out, int N) {
  int n = blockIdx.x * blockDim.x + threadIdx.x;
  if (n >= N) return;
  float acc = bp[0];
#pragma unroll 8
  for (int c = 0; c < 64; ++c) {
    float u = fmaxf(zn[(size_t)n * 64 + c] * stats[128 + c] + stats[192 + c], 0.f);
    acc += u * Wp[c];
  }
  out[n] = acc;
}

// ---------------------------------------------------------------------------
extern "C" void kernel_launch(void* const* d_in, const int* in_sizes, int n_in,
                              void* d_out, int out_size, void* d_ws, size_t ws_size,
                              hipStream_t stream) {
  const float* pos   = (const float*)d_in[0];
  const float* vel   = (const float*)d_in[1];
  const int*   eidx  = (const int*)d_in[2];     // [2,E] (src row, dst row)
  const float* W_in  = (const float*)d_in[3];
  const float* b_in  = (const float*)d_in[4];
  const float* mW1   = (const float*)d_in[5];
  const float* mb1   = (const float*)d_in[6];
  const float* mg1   = (const float*)d_in[7];
  const float* mB1   = (const float*)d_in[8];
  const float* mW2   = (const float*)d_in[9];
  const float* mb2   = (const float*)d_in[10];
  const float* mg2   = (const float*)d_in[11];
  const float* mB2   = (const float*)d_in[12];
  const float* uW1   = (const float*)d_in[13];
  const float* ub1   = (const float*)d_in[14];
  const float* ug1   = (const float*)d_in[15];
  const float* uB1   = (const float*)d_in[16];
  const float* uW2   = (const float*)d_in[17];
  const float* ub2   = (const float*)d_in[18];
  const float* ug2   = (const float*)d_in[19];
  const float* uB2   = (const float*)d_in[20];
  const float* Wpred = (const float*)d_in[21];
  const float* bpred = (const float*)d_in[22];

  const int N = in_sizes[0] / 2;
  const int E = in_sizes[2] / 2;
  const int* esrc = eidx;
  const int* edst = eidx + E;

  // Workspace carve-up (256B aligned blocks).
  char* ws = (char*)d_ws;
  size_t off = 0;
  auto take = [&](size_t bytes) -> void* {
    void* p = ws + off;
    off += (bytes + 255) & ~(size_t)255;
    return p;
  };
  u16*   xu    = (u16*)take((size_t)N * 128 * sizeof(u16));   // [h | aggr] bf16
  u16*   eb1   = (u16*)take((size_t)E * 64 * sizeof(u16));    // edge hidden bf16
  float* ze    = (float*)take((size_t)E * 64 * sizeof(float));// edge pre-activations
  float* aggr  = (float*)take((size_t)N * 64 * sizeof(float));
  float* zn    = (float*)take((size_t)N * 64 * sizeof(float));// node pre-activations
  u16*   nb1   = (u16*)take((size_t)N * 64 * sizeof(u16));    // node hidden bf16
  u16*   mW1b  = (u16*)take(128 * 64 * sizeof(u16));
  u16*   mW2b  = (u16*)take(64 * 64 * sizeof(u16));
  u16*   uW1b  = (u16*)take(128 * 64 * sizeof(u16));
  u16*   uW2b  = (u16*)take(64 * 64 * sizeof(u16));
  float* stats = (float*)take(4 * 256 * sizeof(float));       // 4 BN layers x {sum,sq,scale,shift}

  hipMemsetAsync(aggr, 0, (size_t)N * 64 * sizeof(float), stream);
  hipMemsetAsync(stats, 0, 4 * 256 * sizeof(float), stream);

  // Weight conversion + input embedding.
  k_f2bf<<<(128 * 64 + 255) / 256, 256, 0, stream>>>(mW1, mW1b, 128 * 64);
  k_f2bf<<<(64 * 64 + 255) / 256, 256, 0, stream>>>(mW2, mW2b, 64 * 64);
  k_f2bf<<<(128 * 64 + 255) / 256, 256, 0, stream>>>(uW1, uW1b, 128 * 64);
  k_f2bf<<<(64 * 64 + 255) / 256, 256, 0, stream>>>(uW2, uW2b, 64 * 64);
  k_embed<<<(N * 64 + 255) / 256, 256, 0, stream>>>(pos, vel, W_in, b_in, xu, N);

  const int etiles = E / 16;          // 100000
  const int ntilesN = N / 16;         // 6250
  const size_t Ee = (size_t)E * 64;
  const size_t Ne = (size_t)N * 64;

  // --- Edge MLP layer 1: [E,128] @ mW1 ---
  k_gemm_bn<128, true><<<2048, 256, 0, stream>>>(xu, esrc, edst, mW1b, mb1,
                                                 ze, stats + 0, etiles);
  k_bn_finalize<<<1, 64, 0, stream>>>(stats + 0, mg1, mB1, 1.f / (float)E);
  k_bnrelu_bf16<<<(unsigned)((Ee + 255) / 256), 256, 0, stream>>>(ze, stats + 0, eb1, Ee);

  // --- Edge MLP layer 2: [E,64] @ mW2, then BN+ReLU+segment-sum ---
  k_gemm_bn<64, false><<<2048, 256, 0, stream>>>(eb1, nullptr, nullptr, mW2b, mb2,
                                                 ze, stats + 256, etiles);
  k_bn_finalize<<<1, 64, 0, stream>>>(stats + 256, mg2, mB2, 1.f / (float)E);
  k_bnrelu_scatter<<<(unsigned)((Ee + 255) / 256), 256, 0, stream>>>(ze, stats + 256,
                                                                     edst, aggr, Ee);
  k_aggr2xu<<<(N * 64 + 255) / 256, 256, 0, stream>>>(aggr, xu, N * 64);

  // --- Node MLP layer 1: [N,128] @ uW1 ---
  k_gemm_bn<128, false><<<512, 256, 0, stream>>>(xu, nullptr, nullptr, uW1b, ub1,
                                                 zn, stats + 512, ntilesN);
  k_bn_finalize<<<1, 64, 0, stream>>>(stats + 512, ug1, uB1, 1.f / (float)N);
  k_bnrelu_bf16<<<(unsigned)((Ne + 255) / 256), 256, 0, stream>>>(zn, stats + 512, nb1, Ne);

  // --- Node MLP layer 2: [N,64] @ uW2, then BN+ReLU+prediction ---
  k_gemm_bn<64, false><<<512, 256, 0, stream>>>(nb1, nullptr, nullptr, uW2b, ub2,
                                                zn, stats + 768, ntilesN);
  k_bn_finalize<<<1, 64, 0, stream>>>(stats + 768, ug2, uB2, 1.f / (float)N);
  k_pred<<<(N + 255) / 256, 256, 0, stream>>>(zn, stats + 768, Wpred, bpred,
                                              (float*)d_out, N);
}
